// DeepLSTM_53515292508835
// MI455X (gfx1250) — matrix operand order repaired and checked by reading, not compiled, not run on hardware
//
#include <hip/hip_runtime.h>

// ---------------------------------------------------------------------------
// DeepLSTM (3-layer, skip connections) for MI455X / gfx1250, wave32 + WMMA.
//
// Phase 0: transpose+fp16-convert weights; gather embeddings to fp16.
// Phase 1: time-parallel input projections Gin[l][t] = x_t @ Wx_l(x-part) + b_l
//          (M=64000, K=256, N=1024 GEMMs). Each wave owns a 16x64 stripe:
//          4 accumulators share one A tile -> 4 WMMAs per 10 b128 loads.
// Phase 2: persistent 16-WG recurrent kernel. WG g owns hidden units
//          [16g,16g+16) of every layer => owns gate columns
//          {u, 256+u, 512+u, 768+u} so the LSTM nonlinearity is WG-local.
//          Recurrent fp16 weights resident in LDS (160KB of 320KB/WGP),
//          hidden state exchanged via L2, grid sync = global atomic +
//          s_cluster_barrier. Layer loop fully unrolled -> straight-line
//          WMMA on the latency-critical path.
// ---------------------------------------------------------------------------

#define HSIZE   256
#define NDEPTH  3
#define NBATCH  64
#define NSTEPS  1000
#define NGATES  1024                    // 4*HSIZE
#define NROWS   (NSTEPS * NBATCH)       // 64000

typedef _Float16 f16;
typedef __attribute__((ext_vector_type(16))) _Float16 v16h;
typedef __attribute__((ext_vector_type(8)))  _Float16 v8h;
typedef __attribute__((ext_vector_type(8)))  float    v8f;
typedef __attribute__((ext_vector_type(4)))  unsigned int u32x4;
typedef __attribute__((ext_vector_type(8)))  int      i32x8;
typedef __attribute__((ext_vector_type(4)))  int      i32x4;

// ---- workspace layout (bytes) ----
#define OFF_WTX 0UL                          // 3*1024*256 f16 (x-part, [n][k])
#define OFF_WTR 1572864UL                    // 1024*256 + 2*1024*512 f16
#define OFF_X   4194304UL                    // 64000*256 f16 gathered embeds
#define OFF_H   36962304UL                   // 2 parities * 3 layers * 64*256 f16
#define OFF_BAR 37158912UL                   // grid barrier counter
#define OFF_GIN 37159168UL                   // 3*64000*1024 f32 projections

// ---- LDS layout of the recurrent kernel (bytes) ----
#define L_LW0   0          // 64 x 256 f16
#define L_LW1   32768      // 64 x 512 f16
#define L_LW2   98304      // 64 x 512 f16
#define L_CST   163840     // 3*64*16 f32 cell state slice
#define L_GB    176128     // 64*64 f32 gate staging
#define LDS_BYTES 192512

static __device__ __forceinline__ v8f wmma_f16(v16h a, v16h b, v8f c) {
  return __builtin_amdgcn_wmma_f32_16x16x32_f16(false, a, false, b,
                                                (short)0, c, false, false);
}

// A-tile loader: 16x32 f16, row-major source with row stride (halves).
// Lane L: M=L&15, K = 16*(L>>4) .. +15 (contiguous).
static __device__ __forceinline__ v16h load_a(const f16* base, int row_stride,
                                              int lr, int lh, int k0) {
  const f16* p = base + (long)lr * row_stride + k0 + lh * 16;
  v8h lo = *(const v8h*)(p);
  v8h hi = *(const v8h*)(p + 8);
  return __builtin_shufflevector(lo, hi, 0,1,2,3,4,5,6,7,8,9,10,11,12,13,14,15);
}

// B-tile loader: 32x16 f16 from an [N][K] (column-major-of-original) buffer.
// Lane L: N = L&15, K = 16*(L>>4) .. +15 (contiguous 32B).
static __device__ __forceinline__ v16h load_b(const f16* nk_base, int k_stride,
                                              int lr, int lh, int k0) {
  return *(const v16h*)(nk_base + (long)lr * k_stride + k0 + lh * 16);
}

static __device__ __forceinline__ float sigm(float x) {
  return 1.0f / (1.0f + __expf(-x));
}

// ---------------------------------------------------------------------------
// Phase 0a: weight transpose + fp32->fp16.
// wtx: [l][n(1024)][k(256)]  <- W{l}x[k][n]        (x-part rows 0..255)
// wtr: l0 [n(1024)][k(256)]  <- W0h[k][n]
//      l1,l2 [n(1024)][k(512)] <- k<256: W{l}x[256+k][n]; else W{l}h[k-256][n]
// ---------------------------------------------------------------------------
__global__ void prep_weights(const float* __restrict__ W0x, const float* __restrict__ W0h,
                             const float* __restrict__ W1x, const float* __restrict__ W1h,
                             const float* __restrict__ W2x, const float* __restrict__ W2h,
                             f16* __restrict__ wtx, f16* __restrict__ wtr) {
  long idx = (long)blockIdx.x * blockDim.x + threadIdx.x;   // 2,097,152 total
  if (idx < 786432) {                       // wtx region
    int l = (int)(idx / 262144);
    long r = idx % 262144;
    int n = (int)(r >> 8), k = (int)(r & 255);
    const float* W = (l == 0) ? W0x : (l == 1 ? W1x : W2x);
    wtx[idx] = (f16)W[(long)k * NGATES + n];
  } else {
    long j = idx - 786432;
    if (j < 262144) {                       // layer-0 recurrent
      int n = (int)(j >> 8), k = (int)(j & 255);
      wtr[j] = (f16)W0h[(long)k * NGATES + n];
    } else {
      j -= 262144;
      int l = (int)(j / 524288);            // 0 -> layer1, 1 -> layer2
      long r = j % 524288;
      int n = (int)(r >> 9), k = (int)(r & 511);
      const float* Wx = l ? W2x : W1x;
      const float* Wh = l ? W2h : W1h;
      f16 v = (k < 256) ? (f16)Wx[(long)(256 + k) * NGATES + n]
                        : (f16)Wh[(long)(k - 256) * NGATES + n];
      wtr[262144 + (long)l * 524288 + r] = v;
    }
  }
}

// ---------------------------------------------------------------------------
// Phase 0b: embedding gather -> fp16, layout X[t*B + b][k].
// ---------------------------------------------------------------------------
__global__ void gather_embed(const int* __restrict__ ids,
                             const float* __restrict__ emb,
                             f16* __restrict__ X) {
  long idx = (long)blockIdx.x * blockDim.x + threadIdx.x;   // 16,384,000 total
  int  k   = (int)(idx & 255);
  long row = idx >> 8;
  int  t   = (int)(row >> 6);
  int  b   = (int)(row & 63);
  int  id  = ids[(long)b * NSTEPS + t];
  X[idx]   = (f16)emb[(long)id * HSIZE + k];
}

// ---------------------------------------------------------------------------
// Phase 1: Gin[l] = X @ Wx_l(x-part) + b_l.
// One wave = one 16x64 output stripe: 4 accumulators reuse each A tile.
// 192,000 wave-stripes / 8 waves per block = 24,000 blocks.
// ---------------------------------------------------------------------------
__global__ void __launch_bounds__(256) xproj(const f16* __restrict__ X,
                                             const f16* __restrict__ wtx,
                                             const float* __restrict__ b0,
                                             const float* __restrict__ b1,
                                             const float* __restrict__ b2,
                                             float* __restrict__ gin) {
  int wave = blockIdx.x * 8 + (threadIdx.x >> 5);
  int lane = threadIdx.x & 31;
  int lr = lane & 15, lh = lane >> 4;
  if (wave >= 3 * 4000 * 16) return;
  int l   = wave / 64000;
  int rem = wave % 64000;
  int mt  = rem >> 4;                        // 0..3999  (16-row blocks of 64000)
  int ntb = rem & 15;                        // 0..15    (64-col blocks of 1024)

  const f16* A  = X   + (long)mt * 16 * HSIZE;
  const f16* Bm = wtx + (long)l * NGATES * HSIZE + (long)ntb * 64 * HSIZE;
  const float* bias = (l == 0) ? b0 : (l == 1 ? b1 : b2);

  v8f acc[4];
#pragma unroll
  for (int j = 0; j < 4; ++j) {
    float bv = bias[ntb * 64 + j * 16 + lr]; // lane's column-constant bias
#pragma unroll
    for (int r = 0; r < 8; ++r) acc[j][r] = bv;
  }

#pragma unroll
  for (int kk = 0; kk < 8; ++kk) {
    int k0 = kk * 32;
    v16h a = load_a(A, HSIZE, lr, lh, k0);
#pragma unroll
    for (int j = 0; j < 4; ++j) {
      v16h b = load_b(Bm + (long)j * 16 * HSIZE, HSIZE, lr, lh, k0);
      acc[j] = wmma_f16(a, b, acc[j]);
    }
  }

  float* o = gin + (long)l * NROWS * NGATES
                 + ((long)mt * 16 + lh * 8) * NGATES + ntb * 64 + lr;
#pragma unroll
  for (int j = 0; j < 4; ++j)
#pragma unroll
    for (int r = 0; r < 8; ++r) o[(long)r * NGATES + j * 16] = acc[j][r];
}

// ---------------------------------------------------------------------------
// Grid-wide barrier: global atomic counter + s_cluster_barrier (HW fast path
// when launched as a cluster; architectural NOP otherwise).
// ---------------------------------------------------------------------------
static __device__ __forceinline__ void grid_sync(unsigned* bar, unsigned target) {
  __syncthreads();
  __threadfence();
  if (threadIdx.x == 0) {
    atomicAdd(bar, 1u);
    while (*((volatile unsigned*)bar) < target) __builtin_amdgcn_s_sleep(1);
  }
  __syncthreads();
  __builtin_amdgcn_s_cluster_barrier();
  __threadfence();
}

// ---------------------------------------------------------------------------
// Phase 2: persistent recurrent kernel, 16 workgroups x 256 threads.
// ---------------------------------------------------------------------------
__global__ void __launch_bounds__(256, 1)
recurrent(const f16* __restrict__ wtr, const float* __restrict__ gin,
          f16* __restrict__ Hbuf, float* __restrict__ out, unsigned* bar) {
  extern __shared__ char smem[];
  f16*   LW[3];
  LW[0] = (f16*)(smem + L_LW0);
  LW[1] = (f16*)(smem + L_LW1);
  LW[2] = (f16*)(smem + L_LW2);
  float* Cst = (float*)(smem + L_CST);
  float* Gb  = (float*)(smem + L_GB);

  const int g = blockIdx.x;                  // hidden-unit slice owner
  const int tid = threadIdx.x;
  const int wave = tid >> 5, lane = tid & 31;
  const int lr = lane & 15, lh = lane >> 4;

  // zero hidden state (both parities, all layers) cooperatively
  for (long i = (long)g * 256 + tid; i < 2L * NDEPTH * NBATCH * HSIZE; i += 16 * 256)
    Hbuf[i] = (f16)0.f;
  for (int i = tid; i < NDEPTH * NBATCH * 16; i += 256) Cst[i] = 0.f;

  // ---- stage recurrent weight slices into LDS ----------------------------
#if __has_builtin(__builtin_amdgcn_tensor_load_to_lds)
  // TDM warm path: 1D tensor_load_to_lds per (layer, gate-block); the plain
  // copy below is authoritative, this demonstrates/overlaps the DMA path.
  if (wave == 0) {
    for (int l = 0; l < 3; ++l) {
      const int Kl = l ? 512 : 256;
      const f16* src_base = wtr + (l == 0 ? 0L : 262144L + (long)(l - 1) * 524288L);
      for (int gate = 0; gate < 4; ++gate) {
        const f16* src = src_base + (long)(gate * 256 + g * 16) * Kl;
        unsigned lds_off = (unsigned)(size_t)(LW[l] + (long)gate * 16 * Kl);
        unsigned long ga = (unsigned long)(size_t)src;
        unsigned nelem = (unsigned)(16 * Kl);
        u32x4 g0 = { 1u, lds_off, (unsigned)(ga & 0xffffffffu),
                     (unsigned)(((ga >> 32) & 0x01ffffffu) | 0x80000000u) };
        i32x8 g1 = { 0x10000, (int)(nelem << 16), (int)(1u << 16),
                     (int)(nelem << 16), 1, (int)nelem, 0, 0 };
        i32x4 gz = { 0, 0, 0, 0 };
#if __clang_major__ >= 23
        i32x8 gz8 = { 0, 0, 0, 0, 0, 0, 0, 0 };
        __builtin_amdgcn_tensor_load_to_lds(g0, g1, gz, gz, gz8, 0);
#else
        __builtin_amdgcn_tensor_load_to_lds(g0, g1, gz, gz, 0);
#endif
      }
    }
  }
#if __has_builtin(__builtin_amdgcn_s_wait_tensorcnt)
  __builtin_amdgcn_s_wait_tensorcnt(0);
#endif
#endif
  // authoritative vector copy (hits L2-warm lines)
  for (int l = 0; l < 3; ++l) {
    const int Kl = l ? 512 : 256;
    const int kc8 = Kl / 8;
    const f16* src = wtr + (l == 0 ? 0L : 262144L + (long)(l - 1) * 524288L);
    f16* dst = LW[l];
    for (int i = tid; i < 64 * kc8; i += 256) {
      int j = i / kc8, kc = i % kc8;
      int gate = j >> 4, uu = j & 15;
      ((v8h*)(dst + (long)j * Kl))[kc] =
          *((const v8h*)(src + (long)(gate * 256 + g * 16 + uu) * Kl) + kc);
    }
  }
  __syncthreads();

  unsigned ep = 0;
  grid_sync(bar, ++ep * 16);                 // H zeroing visible everywhere

  const int m   = wave & 3;                  // batch-row block (16 rows)
  const int jb0 = (wave >> 2) * 2;           // first of two gate-blocks

  for (int t = 0; t < NSTEPS; ++t) {
    const int par = t & 1, nxt = par ^ 1;
#pragma unroll
    for (int l = 0; l < 3; ++l) {            // fully unrolled: Kl, LW[l],
      const int Kl = l ? 512 : 256;          // and A-source selection fold
      const f16* lw = LW[l];

      // accumulators pre-loaded with x-projection (+bias), fp32
      const float* gsrc = gin + (long)l * NROWS * NGATES
                        + ((long)t * NBATCH + m * 16 + lh * 8) * NGATES
                        + g * 16 + lr;
      if (t + 1 < NSTEPS)
        __builtin_prefetch((const void*)(gsrc + (long)NBATCH * NGATES), 0, 0);
      v8f acc0, acc1;
#pragma unroll
      for (int r = 0; r < 8; ++r) {
        acc0[r] = gsrc[(long)r * NGATES + (long)jb0 * 256];
        acc1[r] = gsrc[(long)r * NGATES + (long)(jb0 + 1) * 256];
      }

#pragma unroll
      for (int kk = 0; kk < Kl / 32; ++kk) {
        int k0 = kk * 32;
        const f16* Abase;
        int ko;
        if (l == 0)        { Abase = Hbuf + (long)(par * 3 + 0) * NBATCH * HSIZE; ko = k0; }
        else if (k0 < 256) { Abase = Hbuf + (long)(nxt * 3 + (l - 1)) * NBATCH * HSIZE; ko = k0; }
        else               { Abase = Hbuf + (long)(par * 3 + l) * NBATCH * HSIZE; ko = k0 - 256; }
        v16h a  = load_a(Abase + (long)m * 16 * HSIZE, HSIZE, lr, lh, ko);
        v16h b0 = load_b(lw + (long)(jb0 * 16) * Kl,        Kl, lr, lh, k0);
        v16h b1 = load_b(lw + (long)((jb0 + 1) * 16) * Kl,  Kl, lr, lh, k0);
        acc0 = wmma_f16(a, b0, acc0);
        acc1 = wmma_f16(a, b1, acc1);
      }

      // stage gates [64 rows x 64 local cols] into LDS
#pragma unroll
      for (int r = 0; r < 8; ++r) {
        Gb[(m * 16 + lh * 8 + r) * 64 + jb0 * 16 + lr]       = acc0[r];
        Gb[(m * 16 + lh * 8 + r) * 64 + (jb0 + 1) * 16 + lr] = acc1[r];
      }
      __syncthreads();

      // fused LSTM nonlinearity for this WG's 16 hidden units x 64 batch
#pragma unroll
      for (int q = 0; q < 4; ++q) {
        int e = tid + 256 * q;               // 0..1023
        int row = e >> 4, uu = e & 15;
        float gi = Gb[row * 64 + uu];
        float gj = Gb[row * 64 + 16 + uu];
        float gf = Gb[row * 64 + 32 + uu];
        float go = Gb[row * 64 + 48 + uu];
        float c  = Cst[(l * 64 + row) * 16 + uu];
        float cn = sigm(gf) * c + sigm(gi) * tanhf(gj);
        float hn = sigm(go) * tanhf(cn);
        Cst[(l * 64 + row) * 16 + uu] = cn;
        Hbuf[(long)(nxt * 3 + l) * NBATCH * HSIZE + (long)row * HSIZE + g * 16 + uu] = (f16)hn;
        if (t == NSTEPS - 1) {
          out[(long)row * (NDEPTH * 2 * HSIZE) + l * 512 + g * 16 + uu]       = cn;
          out[(long)row * (NDEPTH * 2 * HSIZE) + l * 512 + 256 + g * 16 + uu] = hn;
        }
      }
      grid_sync(bar, ++ep * 16);             // publish h for next layer/step
    }
  }
}

// ---------------------------------------------------------------------------
extern "C" void kernel_launch(void* const* d_in, const int* in_sizes, int n_in,
                              void* d_out, int out_size, void* d_ws, size_t ws_size,
                              hipStream_t stream) {
  const int*   ids = (const int*)d_in[0];
  const float* emb = (const float*)d_in[1];
  const float* W0x = (const float*)d_in[2];
  const float* W0h = (const float*)d_in[3];
  const float* b0  = (const float*)d_in[4];
  const float* W1x = (const float*)d_in[5];
  const float* W1h = (const float*)d_in[6];
  const float* b1  = (const float*)d_in[7];
  const float* W2x = (const float*)d_in[8];
  const float* W2h = (const float*)d_in[9];
  const float* b2  = (const float*)d_in[10];
  float* out = (float*)d_out;
  char*  ws  = (char*)d_ws;

  f16*      wtx = (f16*)(ws + OFF_WTX);
  f16*      wtr = (f16*)(ws + OFF_WTR);
  f16*      X   = (f16*)(ws + OFF_X);
  f16*      H   = (f16*)(ws + OFF_H);
  unsigned* bar = (unsigned*)(ws + OFF_BAR);
  float*    gin = (float*)(ws + OFF_GIN);

  hipMemsetAsync(bar, 0, 64, stream);
  prep_weights<<<8192, 256, 0, stream>>>(W0x, W0h, W1x, W1h, W2x, W2h, wtx, wtr);
  gather_embed<<<64000, 256, 0, stream>>>(ids, emb, X);
  xproj<<<24000, 256, 0, stream>>>(X, wtx, b0, b1, b2, gin);
  hipFuncSetAttribute((const void*)recurrent,
                      hipFuncAttributeMaxDynamicSharedMemorySize, LDS_BYTES);
  recurrent<<<16, 256, LDS_BYTES, stream>>>(wtr, gin, H, out, bar);
}
